// JointAttention_7146825581041
// MI455X (gfx1250) — compile-verified, matrix operand order checked by reading, not run on hardware
//
#include <hip/hip_runtime.h>
#include <hip/hip_bf16.h>
#include <stdint.h>

// ---------------------------------------------------------------------------
// CDNA5 / gfx1250 joint attention. All GEMM-shaped math goes through
// v_wmma_f32_16x16x32_f16 (wave32, 16x16 tiles, f32 accumulate). The GEMM
// A-tile is staged by the Tensor Data Mover (tensor_load_to_lds + TENSORcnt)
// when the builtin is available, with VALU staging as fallback.
// ---------------------------------------------------------------------------

typedef _Float16 f16x8  __attribute__((ext_vector_type(8)));
typedef _Float16 f16x16 __attribute__((ext_vector_type(16)));
typedef float    f32x8  __attribute__((ext_vector_type(8)));
typedef unsigned int u32x4 __attribute__((ext_vector_type(4)));
typedef int i32x4 __attribute__((ext_vector_type(4)));
typedef int i32x8 __attribute__((ext_vector_type(8)));

#if defined(__has_builtin)
#if __has_builtin(__builtin_amdgcn_tensor_load_to_lds) && \
    __has_builtin(__builtin_amdgcn_s_wait_tensorcnt)
#define HAVE_TDM 1
#endif
#endif

constexpr int BATCH = 2;
constexpr int NDIM  = 512;
constexpr int NHEADS = 8;
constexpr int NKVH  = 2;
constexpr int NDH   = 64;
constexpr int NSEG  = 1024;
constexpr int NSEGS = 8;
constexpr int TNA   = 4096;
constexpr int TNX   = 4096;
constexpr int NTOK  = TNA + TNX;          // 8192 tokens per batch (a ++ x)
constexpr size_t OUTX_ELEMS = (size_t)BATCH * TNX * NHEADS * NDH; // out_x region

#define DEV static __device__ __forceinline__

DEV f16x16 join8(f16x8 lo, f16x8 hi) {
  f16x16 r;
#pragma unroll
  for (int i = 0; i < 8; ++i) { r[i] = lo[i]; r[i + 8] = hi[i]; }
  return r;
}

// Loads one 16x32 (A) or 32x16 (B) f16 WMMA fragment. Per the CDNA5 ISA
// 16-bit layout, lane (hi = lane>>4) element e maps to K = (e<8?e:e+8)+8*hi,
// i.e. two contiguous 16-byte chunks at +0 and +16 halves. Caller passes
// p = row_base + hi*8 (halves); row/col selection is done by the caller.
DEV f16x16 load_frag(const _Float16* p) {
  return join8(*(const f16x8*)p, *(const f16x8*)(p + 16));
}

DEV f32x8 wmma16(f16x16 a, f16x16 b, f32x8 c) {
  return __builtin_amdgcn_wmma_f32_16x16x32_f16(false, a, false, b, (short)0, c,
                                                false, false);
}

// ---------------------------------------------------------------------------
// Kernel 0: f32 -> f16 convert (inputs + weights)
// ---------------------------------------------------------------------------
__global__ __launch_bounds__(256) void cvt_f32_f16(const float* __restrict__ src,
                                                   _Float16* __restrict__ dst,
                                                   int n) {
  int i = blockIdx.x * 256 + threadIdx.x;
  if (i < n) dst[i] = (_Float16)src[i];
}

// ---------------------------------------------------------------------------
// Kernel 1: projection GEMM. Y[b, tok_off+t, :] = X[row,:] @ W
// X: [B*4096, 512] f16 (row-linear), W: [512, CO] f16, Y: f32 ld = co_total.
// Block tile 128x64, 8 waves of 32x32 (2x2 WMMA accums), K-step 32.
// A tile staged by TDM (pad_interval=16dw, pad_amount=4dw matches the +8-half
// LDS row padding); W tile staged transposed by VALU.
// ---------------------------------------------------------------------------
__global__ __launch_bounds__(256) void gemm_f16_wmma(
    const _Float16* __restrict__ X, const _Float16* __restrict__ W,
    float* __restrict__ Y, int CO, int co_total, int tok_off) {
  __shared__ __align__(16) _Float16 As[128][40];  // 32 data + 8 pad halves
  __shared__ __align__(16) _Float16 Bs[64][40];   // W tile transposed [n][k]

  const int tiles_n = CO / 64;
  const int tm = blockIdx.x / tiles_n;
  const int tn = blockIdx.x % tiles_n;
  const int row0 = tm * 128;
  const int col0 = tn * 64;
  const int tid  = threadIdx.x;
  const int lane = tid & 31;
  const int wave = tid >> 5;
  const int wm = wave & 3;   // 4 waves along M (32 rows each)
  const int wn = wave >> 2;  // 2 waves along N (32 cols each)
  const int hi = lane >> 4;
  const int lr = lane & 15;

#ifdef HAVE_TDM
  const unsigned int lds_base = (unsigned int)(uintptr_t)(&As[0][0]);
  const unsigned long long gbase =
      (unsigned long long)(uintptr_t)X + (unsigned long long)row0 * (NDIM * 2);
#endif

  f32x8 acc[2][2] = {};

  for (int k0 = 0; k0 < NDIM; k0 += 32) {
    __syncthreads();
#ifdef HAVE_TDM
    if (wave == 0) {
      // Tensor DMA descriptor: 2D tile 32(x) x 128(y) of 2-byte elements.
      unsigned long long ga = gbase + (unsigned long long)k0 * 2ull;
      u32x4 g0;
      g0[0] = 1u;                      // count=1 (one valid descriptor)
      g0[1] = lds_base;                // lds_addr (bytes)
      g0[2] = (unsigned int)ga;        // global_addr[31:0]
      g0[3] = (unsigned int)((ga >> 32) & 0x01FFFFFFu) | (2u << 30);  // +type=2
      i32x8 g1;
      g1[0] = (1 << 16)    // data_size = 2 bytes
            | (1 << 20)    // pad_enable
            | (3 << 22)    // pad_interval: 16 DWORDs (= 32 halves per row)
            | (3 << 25);   // pad_amount: 4 DWORDs (= 8 pad halves)
      g1[1] = (int)(512u << 16);    // tensor_dim0 = 512 (bits 79:48)
      g1[2] = (int)(8192u << 16);   // tensor_dim1 = 8192 (bits 111:80)
      g1[3] = (int)(32u << 16);     // tile_dim0 = 32 (bits 127:112)
      g1[4] = 128;                  // tile_dim1 = 128 (bits 143:128)
      g1[5] = 512;                  // tensor_dim0_stride = 512 (bits 207:160)
      g1[6] = 0;
      g1[7] = 0;
      i32x4 g2 = {0, 0, 0, 0};
      i32x4 g3 = {0, 0, 0, 0};
#if defined(__clang_major__) && __clang_major__ >= 23
      i32x8 g4 = {0, 0, 0, 0, 0, 0, 0, 0};
      __builtin_amdgcn_tensor_load_to_lds(g0, g1, g2, g3, g4, 0);
#else
      __builtin_amdgcn_tensor_load_to_lds(g0, g1, g2, g3, 0);
#endif
    }
#else
    {  // stage A: 128x32 halves, 16 halves/thread (two b128 chunks)
      int idx = tid * 16;
      int r = idx >> 5, c = idx & 31;
      const f16x8* src = (const f16x8*)(X + (size_t)(row0 + r) * NDIM + k0 + c);
      *(f16x8*)&As[r][c]     = src[0];
      *(f16x8*)&As[r][c + 8] = src[1];
    }
#endif
    {  // stage B transposed: Bs[n][k] = W[k0+k][col0+n]
      int idx = tid * 8;
      int k = idx >> 6, n = idx & 63;
      f16x8 v = *(const f16x8*)(W + (size_t)(k0 + k) * CO + col0 + n);
#pragma unroll
      for (int i = 0; i < 8; ++i) Bs[n + i][k] = v[i];
    }
#ifdef HAVE_TDM
    if (wave == 0) __builtin_amdgcn_s_wait_tensorcnt(0);
#endif
    __syncthreads();

    if (k0 + 32 < NDIM)  // gfx1250 global_prefetch_b8 on next A tile
      __builtin_prefetch(X + (size_t)(row0 + (tid >> 1)) * NDIM + k0 + 32 +
                             (tid & 1) * 16, 0, 1);

    f16x16 a[2], bfr[2];
#pragma unroll
    for (int i = 0; i < 2; ++i)
      a[i] = load_frag(&As[wm * 32 + i * 16 + lr][hi * 8]);
#pragma unroll
    for (int j = 0; j < 2; ++j)
      bfr[j] = load_frag(&Bs[wn * 32 + j * 16 + lr][hi * 8]);
#pragma unroll
    for (int i = 0; i < 2; ++i)
#pragma unroll
      for (int j = 0; j < 2; ++j) acc[i][j] = wmma16(a[i], bfr[j], acc[i][j]);
  }

  // epilogue: C layout lane holds (n = lr, m = r + hi*8); streamed -> NT store
#pragma unroll
  for (int i = 0; i < 2; ++i) {
#pragma unroll
    for (int j = 0; j < 2; ++j) {
      int col = col0 + wn * 32 + j * 16 + lr;
#pragma unroll
      for (int r = 0; r < 8; ++r) {
        int gr = row0 + wm * 32 + i * 16 + hi * 8 + r;  // row in [0, B*4096)
        int bb = gr >> 12, t = gr & 4095;
        __builtin_nontemporal_store(
            acc[i][j][r],
            &Y[((size_t)bb * NTOK + tok_off + t) * co_total + col]);
      }
    }
  }
}

// ---------------------------------------------------------------------------
// Kernel 2: per-head RMSNorm + RoPE, elu feature map, f16 repack.
// Row ids: [0, B*N*H)            -> q rows
//          [.., +B*N*KVH)        -> k rows
//          [.., +B*N*KVH)        -> v rows (convert only)
// ---------------------------------------------------------------------------
__global__ __launch_bounds__(256) void norm_rope(
    const float* __restrict__ qf, const float* __restrict__ kf,
    const float* __restrict__ vf, const float* __restrict__ gq_x,
    const float* __restrict__ gk_x, const float* __restrict__ gq_a,
    const float* __restrict__ gk_a, _Float16* __restrict__ q16,
    _Float16* __restrict__ sq16, _Float16* __restrict__ k16,
    _Float16* __restrict__ sk16, _Float16* __restrict__ v16) {
  const int qrows = BATCH * NTOK * NHEADS;
  const int krows = BATCH * NTOK * NKVH;
  int idx = blockIdx.x * 256 + threadIdx.x;
  if (idx >= qrows + 2 * krows) return;

  float v[64];
  if (idx >= qrows + krows) {  // v: straight convert
    int r = idx - qrows - krows;
    int kv = r % NKVH;  r /= NKVH;
    int t = r % NTOK;   int b = r / NTOK;
    const float* src = vf + ((size_t)b * NTOK + t) * (NKVH * NDH) + kv * NDH;
    _Float16* dst = v16 + ((size_t)(b * NKVH + kv) * NTOK + t) * NDH;
    for (int d = 0; d < 64; ++d) dst[d] = (_Float16)src[d];
    return;
  }

  const float* src; const float* g; _Float16 *o16, *so16; int t;
  if (idx < qrows) {
    int r = idx;
    int h = r % NHEADS; r /= NHEADS;
    t = r % NTOK; int b = r / NTOK;
    src = qf + ((size_t)b * NTOK + t) * (NHEADS * NDH) + h * NDH;
    g   = ((t < TNA) ? gq_a : gq_x) + h * NDH;
    o16  = q16  + ((size_t)(b * NHEADS + h) * NTOK + t) * NDH;
    so16 = sq16 + ((size_t)(b * NHEADS + h) * NTOK + t) * NDH;
  } else {
    int r = idx - qrows;
    int kv = r % NKVH; r /= NKVH;
    t = r % NTOK; int b = r / NTOK;
    src = kf + ((size_t)b * NTOK + t) * (NKVH * NDH) + kv * NDH;
    g   = ((t < TNA) ? gk_a : gk_x) + kv * NDH;
    o16  = k16  + ((size_t)(b * NKVH + kv) * NTOK + t) * NDH;
    so16 = sk16 + ((size_t)(b * NKVH + kv) * NTOK + t) * NDH;
  }

  float ss = 0.f;
  for (int d = 0; d < 64; ++d) { v[d] = src[d]; ss += v[d] * v[d]; }
  float sc = 8.0f / fmaxf(sqrtf(ss), 1e-12f);  // sqrt(DH)/max(||v||,eps)
  for (int d = 0; d < 64; ++d) v[d] *= sc * g[d];

  // RoPE, pos = global token index
  float o[64];
  const float l4 = logf(10000.0f) * (1.0f / 32.0f);
  for (int i = 0; i < 32; ++i) {
    float ang = (float)t * expf(-(float)i * l4);
    float c = cosf(ang), s = sinf(ang);
    o[i]      = v[i] * c - v[i + 32] * s;
    o[i + 32] = v[i + 32] * c + v[i] * s;
  }
  for (int d = 0; d < 64; ++d) {
    float x = o[d];
    o16[d]  = (_Float16)x;
    so16[d] = (_Float16)(x > 0.f ? x + 1.f : expf(x));  // elu(x)+1
  }
}

// ---------------------------------------------------------------------------
// Kernel 3: per-segment memory contribution dM[d][e] = sum_t sk[t][d]*v[t][e]
// One block per (b,kv,seg); A=sk^T and B=V staged transposed into LDS so both
// fragments are contiguous ds_load_b128.
// ---------------------------------------------------------------------------
__global__ __launch_bounds__(256) void seg_kv(const _Float16* __restrict__ sk16,
                                              const _Float16* __restrict__ v16,
                                              float* __restrict__ dM) {
  __shared__ __align__(16) _Float16 skT[64][40];
  __shared__ __align__(16) _Float16 vT[64][40];
  const int s = blockIdx.x % NSEGS;
  const int kvb = blockIdx.x / NSEGS;  // b*KVH + kv
  const _Float16* skp = sk16 + ((size_t)kvb * NTOK + s * NSEG) * NDH;
  const _Float16* vp  = v16  + ((size_t)kvb * NTOK + s * NSEG) * NDH;

  const int tid = threadIdx.x, lane = tid & 31, wave = tid >> 5;
  const int hi = lane >> 4, lr = lane & 15;
  const int d0 = (wave & 3) * 16;   // 4 row tiles
  const int e0 = (wave >> 2) * 32;  // 2 col groups of 32

  f32x8 acc[2] = {};
  for (int kc = 0; kc < NSEG; kc += 32) {
    __syncthreads();
    {  // transpose-stage 32 tokens x 64 dims of sk and v
      int t = tid >> 3, d = (tid & 7) * 8;
      f16x8 sv = *(const f16x8*)(skp + (size_t)(kc + t) * NDH + d);
      f16x8 vv = *(const f16x8*)(vp  + (size_t)(kc + t) * NDH + d);
#pragma unroll
      for (int i = 0; i < 8; ++i) { skT[d + i][t] = sv[i]; vT[d + i][t] = vv[i]; }
    }
    __syncthreads();
    f16x16 a = load_frag(&skT[d0 + lr][hi * 8]);
#pragma unroll
    for (int j = 0; j < 2; ++j) {
      f16x16 b = load_frag(&vT[e0 + j * 16 + lr][hi * 8]);
      acc[j] = wmma16(a, b, acc[j]);
    }
  }
  float* out = dM + ((size_t)kvb * NSEGS + s) * (NDH * NDH);
#pragma unroll
  for (int j = 0; j < 2; ++j)
#pragma unroll
    for (int r = 0; r < 8; ++r)
      out[(d0 + hi * 8 + r) * NDH + e0 + j * 16 + lr] = acc[j][r];
}

// ---------------------------------------------------------------------------
// Kernel 4: exclusive prefix over segments for M (stored transposed, f16) and
// z (f32). One block per (b,kv).
// ---------------------------------------------------------------------------
__global__ __launch_bounds__(256) void seg_prefix(
    const _Float16* __restrict__ sk16, const float* __restrict__ dM,
    _Float16* __restrict__ MpreT, float* __restrict__ zpre) {
  __shared__ float zseg[NSEGS][64];
  const int kvb = blockIdx.x;
  const _Float16* skp = sk16 + (size_t)kvb * NTOK * NDH;

  for (int sd = threadIdx.x; sd < NSEGS * 64; sd += 256) {
    int s = sd >> 6, d = sd & 63;
    const _Float16* p = skp + (size_t)s * NSEG * NDH + d;
    float acc = 0.f;
    for (int t = 0; t < NSEG; ++t) acc += (float)p[t * NDH];
    zseg[s][d] = acc;
  }
  __syncthreads();
  if (threadIdx.x < 64) {
    int d = threadIdx.x; float acc = 0.f;
    for (int s = 0; s < NSEGS; ++s) {
      zpre[((size_t)kvb * NSEGS + s) * 64 + d] = acc;
      acc += zseg[s][d];
    }
  }
  const float* dMp = dM + (size_t)kvb * NSEGS * (NDH * NDH);
  for (int i = threadIdx.x; i < NDH * NDH; i += 256) {
    int d = i >> 6, e = i & 63;
    float acc = 0.f;
    for (int s = 0; s < NSEGS; ++s) {
      MpreT[((size_t)kvb * NSEGS + s) * (NDH * NDH) + e * NDH + d] = (_Float16)acc;
      acc += dMp[(size_t)s * (NDH * NDH) + i];
    }
  }
}

// ---------------------------------------------------------------------------
// Kernel 5: fused attention. One block = 128 queries of one (b,h,seg); 8 waves
// each own 16 query rows. Flash-style online softmax over 64-key chunks +
// linear-memory readout + gate + scatter to out_x/out_a layout.
// ---------------------------------------------------------------------------
__global__ __launch_bounds__(256) void attn(
    const _Float16* __restrict__ q16, const _Float16* __restrict__ sq16,
    const _Float16* __restrict__ k16, const _Float16* __restrict__ v16,
    const _Float16* __restrict__ MpreT, const float* __restrict__ zpre,
    const float* __restrict__ beta, float* __restrict__ out) {
  __shared__ __align__(16) _Float16 vT[64][72];
  __shared__ __align__(16) _Float16 Pb[8][16][72];

  int x = blockIdx.x;
  const int qt = x & 7; x >>= 3;   // 8 query tiles of 128 per segment
  const int s  = x & 7; x >>= 3;
  const int h  = x & 7; x >>= 3;
  const int b  = x;
  const int kv = h % NKVH;         // gqa tile order: heads = [kv0,kv1,...]
  const float gate = 1.0f / (1.0f + expf(-beta[h]));

  const _Float16* qp  = q16  + ((size_t)(b * NHEADS + h) * NTOK + s * NSEG + qt * 128) * NDH;
  const _Float16* sqp = sq16 + ((size_t)(b * NHEADS + h) * NTOK + s * NSEG + qt * 128) * NDH;
  const _Float16* kp  = k16  + ((size_t)(b * NKVH + kv) * NTOK + s * NSEG) * NDH;
  const _Float16* vp  = v16  + ((size_t)(b * NKVH + kv) * NTOK + s * NSEG) * NDH;
  const _Float16* Mp  = MpreT + ((size_t)(b * NKVH + kv) * NSEGS + s) * (NDH * NDH);
  const float*    zp  = zpre  + ((size_t)(b * NKVH + kv) * NSEGS + s) * NDH;

  const int tid = threadIdx.x, lane = tid & 31, wave = tid >> 5;
  const int hi = lane >> 4, lr = lane & 15;
  const int mrow0 = wave * 16;  // query rows owned by this wave

  // Q / sq fragments (reused across all chunks)
  f16x16 qa[2], sqa[2];
#pragma unroll
  for (int ks = 0; ks < 2; ++ks) {
    qa[ks]  = load_frag(qp  + (size_t)(mrow0 + lr) * NDH + ks * 32 + hi * 8);
    sqa[ks] = load_frag(sqp + (size_t)(mrow0 + lr) * NDH + ks * 32 + hi * 8);
  }

  // ---- memory readout: a_mem = sq @ Mpre, den = sq . z + eps -------------
  f32x8 am[4] = {};
#pragma unroll
  for (int nt = 0; nt < 4; ++nt)
#pragma unroll
    for (int ks = 0; ks < 2; ++ks) {
      f16x16 bf = load_frag(Mp + (size_t)(nt * 16 + lr) * NDH + ks * 32 + hi * 8);
      am[nt] = wmma16(sqa[ks], bf, am[nt]);
    }
  float den;
  {
    const _Float16* sp = sqp + (size_t)(mrow0 + lr) * NDH;
    float acc = 0.f;
    for (int d = 0; d < 64; ++d) acc += (float)sp[d] * zp[d];
    den = acc + 1e-12f;
  }
  float god[8];  // gate / den per accumulator row (row = r + hi*8)
#pragma unroll
  for (int r = 0; r < 8; ++r) god[r] = gate / __shfl(den, r + hi * 8, 32);
#pragma unroll
  for (int nt = 0; nt < 4; ++nt)
#pragma unroll
    for (int r = 0; r < 8; ++r) am[nt][r] *= god[r];

  // ---- local causal softmax attention (flash over 64-key chunks) ---------
  f32x8 ao[4] = {};
  float rmax[8], rsum[8];
#pragma unroll
  for (int r = 0; r < 8; ++r) { rmax[r] = -3.0e38f; rsum[r] = 0.f; }
  const int qbase_w = qt * 128 + mrow0;
  const int nchunks = 2 * qt + 2;  // keys 0 .. qt*128+127

  for (int c = 0; c < nchunks; ++c) {
    __syncthreads();
    {  // stage V chunk transposed: vT[dh][key]
      int t = tid >> 2, d0 = (tid & 3) * 16;
      const f16x8* src = (const f16x8*)(vp + (size_t)(c * 64 + t) * NDH + d0);
      f16x8 v0 = src[0], v1 = src[1];
#pragma unroll
      for (int i = 0; i < 8; ++i) { vT[d0 + i][t] = v0[i]; vT[d0 + 8 + i][t] = v1[i]; }
    }
    __syncthreads();

    // scores S = Q K^T * scale (B operand straight from global: K row-major)
    f32x8 sc[4] = {};
#pragma unroll
    for (int nt = 0; nt < 4; ++nt)
#pragma unroll
      for (int ks = 0; ks < 2; ++ks) {
        f16x16 bf = load_frag(kp + (size_t)(c * 64 + nt * 16 + lr) * NDH +
                              ks * 32 + hi * 8);
        sc[nt] = wmma16(qa[ks], bf, sc[nt]);
      }
    // scale + causal mask (segment-local)
#pragma unroll
    for (int nt = 0; nt < 4; ++nt)
#pragma unroll
      for (int r = 0; r < 8; ++r) {
        float v = sc[nt][r] * 0.125f;               // 64^-0.5
        int key = c * 64 + nt * 16 + lr;
        int qrow = qbase_w + hi * 8 + r;
        sc[nt][r] = (key > qrow) ? -1.0e30f : v;
      }
    // online softmax stats (row lives in the 16-lane half; xor-reduce)
    float cmax[8];
#pragma unroll
    for (int r = 0; r < 8; ++r) {
      float m = sc[0][r];
#pragma unroll
      for (int nt = 1; nt < 4; ++nt) m = fmaxf(m, sc[nt][r]);
#pragma unroll
      for (int msk = 1; msk < 16; msk <<= 1) m = fmaxf(m, __shfl_xor(m, msk, 32));
      cmax[r] = m;
    }
#pragma unroll
    for (int r = 0; r < 8; ++r) {
      float nm = fmaxf(rmax[r], cmax[r]);
      float corr = expf(rmax[r] - nm);
      rmax[r] = nm; rsum[r] *= corr;
#pragma unroll
      for (int nt = 0; nt < 4; ++nt) ao[nt][r] *= corr;
    }
    float psum[8];
#pragma unroll
    for (int r = 0; r < 8; ++r) psum[r] = 0.f;
#pragma unroll
    for (int nt = 0; nt < 4; ++nt)
#pragma unroll
      for (int r = 0; r < 8; ++r) {
        float p = expf(sc[nt][r] - rmax[r]);
        sc[nt][r] = p; psum[r] += p;
      }
#pragma unroll
    for (int r = 0; r < 8; ++r) {
      float p = psum[r];
#pragma unroll
      for (int msk = 1; msk < 16; msk <<= 1) p += __shfl_xor(p, msk, 32);
      rsum[r] += p;
    }
    // C-layout -> A-layout through LDS, then P @ V
#pragma unroll
    for (int nt = 0; nt < 4; ++nt)
#pragma unroll
      for (int r = 0; r < 8; ++r)
        Pb[wave][hi * 8 + r][nt * 16 + lr] = (_Float16)sc[nt][r];
    __syncthreads();
    f16x16 pf[2];
#pragma unroll
    for (int ks = 0; ks < 2; ++ks)
      pf[ks] = load_frag(&Pb[wave][lr][ks * 32 + hi * 8]);
#pragma unroll
    for (int nt = 0; nt < 4; ++nt)
#pragma unroll
      for (int ks = 0; ks < 2; ++ks) {
        f16x16 bf = load_frag(&vT[nt * 16 + lr][ks * 32 + hi * 8]);
        ao[nt] = wmma16(pf[ks], bf, ao[nt]);
      }
  }

  // ---- combine + scatter to (out_x ++ out_a) layout ----------------------
  float inv_rs[8];
#pragma unroll
  for (int r = 0; r < 8; ++r) inv_rs[r] = 1.0f / rsum[r];
  const float og = 1.0f - gate;
#pragma unroll
  for (int nt = 0; nt < 4; ++nt)
#pragma unroll
    for (int r = 0; r < 8; ++r) {
      float val = am[nt][r] + og * ao[nt][r] * inv_rs[r];
      int mseg = qt * 128 + mrow0 + hi * 8 + r;
      int tg = s * NSEG + mseg;  // global token in [a ++ x]
      int dcol = h * NDH + nt * 16 + lr;
      size_t addr;
      if (tg < TNA)  // 'a' tokens -> out_a region (after out_x)
        addr = OUTX_ELEMS + ((size_t)(b * TNA + tg)) * (NHEADS * NDH) + dcol;
      else
        addr = ((size_t)(b * TNX + (tg - TNA))) * (NHEADS * NDH) + dcol;
      __builtin_nontemporal_store(val, &out[addr]);
    }
}

// ---------------------------------------------------------------------------
// Host launcher
// ---------------------------------------------------------------------------
extern "C" void kernel_launch(void* const* d_in, const int* in_sizes, int n_in,
                              void* d_out, int out_size, void* d_ws,
                              size_t ws_size, hipStream_t stream) {
  (void)in_sizes; (void)n_in; (void)out_size; (void)ws_size;
  const float* x    = (const float*)d_in[0];
  const float* a    = (const float*)d_in[1];
  const float* Wq_x = (const float*)d_in[2];
  const float* Wk_x = (const float*)d_in[3];
  const float* Wv_x = (const float*)d_in[4];
  const float* Wq_a = (const float*)d_in[5];
  const float* Wk_a = (const float*)d_in[6];
  const float* Wv_a = (const float*)d_in[7];
  const float* gq_x = (const float*)d_in[8];
  const float* gk_x = (const float*)d_in[9];
  const float* gq_a = (const float*)d_in[10];
  const float* gk_a = (const float*)d_in[11];
  const float* beta = (const float*)d_in[12];
  float* out = (float*)d_out;

  char* ws = (char*)d_ws;
  size_t off = 0;
  auto alloc = [&](size_t bytes) {
    void* p = ws + off;
    off = (off + bytes + 255) & ~(size_t)255;
    return p;
  };
  const size_t SRC = (size_t)BATCH * TNA * NDIM;  // elements per source
  _Float16* a16   = (_Float16*)alloc(SRC * 2);
  _Float16* x16   = (_Float16*)alloc(SRC * 2);
  _Float16* wqa16 = (_Float16*)alloc((size_t)NDIM * 512 * 2);
  _Float16* wka16 = (_Float16*)alloc((size_t)NDIM * 128 * 2);
  _Float16* wva16 = (_Float16*)alloc((size_t)NDIM * 128 * 2);
  _Float16* wqx16 = (_Float16*)alloc((size_t)NDIM * 512 * 2);
  _Float16* wkx16 = (_Float16*)alloc((size_t)NDIM * 128 * 2);
  _Float16* wvx16 = (_Float16*)alloc((size_t)NDIM * 128 * 2);
  float* qf = (float*)alloc((size_t)BATCH * NTOK * 512 * 4);
  float* kf = (float*)alloc((size_t)BATCH * NTOK * 128 * 4);
  float* vf = (float*)alloc((size_t)BATCH * NTOK * 128 * 4);
  _Float16* q16  = (_Float16*)alloc((size_t)BATCH * NHEADS * NTOK * NDH * 2);
  _Float16* sq16 = (_Float16*)alloc((size_t)BATCH * NHEADS * NTOK * NDH * 2);
  _Float16* k16  = (_Float16*)alloc((size_t)BATCH * NKVH * NTOK * NDH * 2);
  _Float16* sk16 = (_Float16*)alloc((size_t)BATCH * NKVH * NTOK * NDH * 2);
  _Float16* v16  = (_Float16*)alloc((size_t)BATCH * NKVH * NTOK * NDH * 2);
  float* dM = (float*)alloc((size_t)BATCH * NKVH * NSEGS * NDH * NDH * 4);
  _Float16* MpreT = (_Float16*)alloc((size_t)BATCH * NKVH * NSEGS * NDH * NDH * 2);
  float* zpre = (float*)alloc((size_t)BATCH * NKVH * NSEGS * NDH * 4);

  auto cvt = [&](const float* s, _Float16* d, size_t n) {
    cvt_f32_f16<<<dim3((unsigned)((n + 255) / 256)), dim3(256), 0, stream>>>(s, d, (int)n);
  };
  cvt(a, a16, SRC);
  cvt(x, x16, SRC);
  cvt(Wq_a, wqa16, (size_t)NDIM * 512);
  cvt(Wk_a, wka16, (size_t)NDIM * 128);
  cvt(Wv_a, wva16, (size_t)NDIM * 128);
  cvt(Wq_x, wqx16, (size_t)NDIM * 512);
  cvt(Wk_x, wkx16, (size_t)NDIM * 128);
  cvt(Wv_x, wvx16, (size_t)NDIM * 128);

  // projections (rows = B*4096 per source -> 64 M-tiles of 128)
  gemm_f16_wmma<<<dim3(64 * 8), dim3(256), 0, stream>>>(a16, wqa16, qf, 512, 512, 0);
  gemm_f16_wmma<<<dim3(64 * 8), dim3(256), 0, stream>>>(x16, wqx16, qf, 512, 512, TNA);
  gemm_f16_wmma<<<dim3(64 * 2), dim3(256), 0, stream>>>(a16, wka16, kf, 128, 128, 0);
  gemm_f16_wmma<<<dim3(64 * 2), dim3(256), 0, stream>>>(x16, wkx16, kf, 128, 128, TNA);
  gemm_f16_wmma<<<dim3(64 * 2), dim3(256), 0, stream>>>(a16, wva16, vf, 128, 128, 0);
  gemm_f16_wmma<<<dim3(64 * 2), dim3(256), 0, stream>>>(x16, wvx16, vf, 128, 128, TNA);

  {
    int rows = BATCH * NTOK * NHEADS + 2 * BATCH * NTOK * NKVH;  // 196608
    norm_rope<<<dim3((rows + 255) / 256), dim3(256), 0, stream>>>(
        qf, kf, vf, gq_x, gk_x, gq_a, gk_a, q16, sq16, k16, sk16, v16);
  }
  seg_kv<<<dim3(BATCH * NKVH * NSEGS), dim3(256), 0, stream>>>(sk16, v16, dM);
  seg_prefix<<<dim3(BATCH * NKVH), dim3(256), 0, stream>>>(sk16, dM, MpreT, zpre);
  attn<<<dim3(BATCH * NHEADS * NSEGS * 8), dim3(256), 0, stream>>>(
      q16, sq16, k16, v16, MpreT, zpre, beta, out);
}
